// MuRILBiLSTMModel_17480516895253
// MI455X (gfx1250) — matrix operand order repaired and checked by reading, not compiled, not run on hardware
//
#include <hip/hip_runtime.h>

// ---------------------------------------------------------------------------
// Model dims
// ---------------------------------------------------------------------------
#define BB   16
#define SS   256
#define TT   64
#define INP  128
#define HD   512
#define G4   2048      // 4*H
#define EMB  128
#define POS  32
#define VOC  32000
#define XIN  640       // EMB + H

// ---------------------------------------------------------------------------
// WMMA types / helpers (CDNA5, wave32, 16x16x32 bf16 -> f32)
// ---------------------------------------------------------------------------
typedef __attribute__((ext_vector_type(16))) __bf16 v16bf;
typedef __attribute__((ext_vector_type(8)))  float  v8f;

union FragU { v16bf v; uint4 q[2]; };

__device__ __forceinline__ __bf16 f2bf(float f) {
    union { float f; unsigned u; } v; v.f = f;
    unsigned r = v.u + 0x7FFFu + ((v.u >> 16) & 1u);   // round-to-nearest-even
    unsigned short h = (unsigned short)(r >> 16);
    union { unsigned short s; __bf16 b; } o; o.s = h;
    return o.b;
}

// Hardware tanh (V_TANH_F32 on gfx1250) when the builtin exists.
#if __has_builtin(__builtin_amdgcn_tanhf)
#define FAST_TANH(x) __builtin_amdgcn_tanhf(x)
#elif __has_builtin(__builtin_amdgcn_tanh_f32)
#define FAST_TANH(x) __builtin_amdgcn_tanh_f32(x)
#else
#define FAST_TANH(x) tanhf(x)
#endif

__device__ __forceinline__ float sigm(float x) {
    return __builtin_amdgcn_rcpf(1.f + __expf(-x));
}

// A-matrix 16x32 bf16 fragment (row-major source, leading dim ld).
// lane<16: row=lane, K[0:8) + K[16:24); lane>=16: row=lane-16, K[8:16) + K[24:32)
__device__ __forceinline__ v16bf load_frag_A(const __bf16* base, int ld, int lane) {
    const __bf16* p = base + (lane & 15) * ld + ((lane >> 4) << 3);
    FragU f;
    f.q[0] = *(const uint4*)(p);
    f.q[1] = *(const uint4*)(p + 16);
    return f.v;
}

// B-matrix 32x16 bf16 fragment from weight W (N x K row-major).
// wp points at W + n0*ld + kb. lane<16: col=lane, K[0:8) + K[16:24); lane>=16: col=lane-16, K[8:16) + K[24:32)
__device__ __forceinline__ v16bf load_frag_B(const __bf16* wp, int ld, int lane) {
    const __bf16* p = wp + (lane & 15) * ld + ((lane >> 4) << 3);
    FragU f;
    f.q[0] = *(const uint4*)(p);
    f.q[1] = *(const uint4*)(p + 16);
    return f.v;
}

__device__ __forceinline__ v8f wmma_bf16(v16bf a, v16bf b, v8f c) {
    return __builtin_amdgcn_wmma_f32_16x16x32_bf16(false, a, false, b, (short)0, c, false, false);
}

__device__ __forceinline__ v8f vzero8() {
    v8f z = {0.f,0.f,0.f,0.f,0.f,0.f,0.f,0.f};
    return z;
}

// ---------------------------------------------------------------------------
// Small utility kernels
// ---------------------------------------------------------------------------
__global__ void cvt_bf16_kernel(const float* __restrict__ in, __bf16* __restrict__ out, int n) {
    int i = blockIdx.x * 256 + threadIdx.x;
    if (i < n) out[i] = f2bf(in[i]);
}

__global__ void bias_sum_kernel(const float* a, const float* b, float* o, int n) {
    int i = blockIdx.x * 256 + threadIdx.x;
    if (i < n) o[i] = a[i] + b[i];
}

// gate = sigmoid(pos_onehot @ attn_Wp + attn_bp)   -> (B,S)
__global__ void gate_ps_kernel(const float* __restrict__ onehot, const float* __restrict__ Wp,
                               const float* __restrict__ bp, float* __restrict__ out) {
    int i = blockIdx.x * 256 + threadIdx.x;
    if (i >= BB * SS) return;
    const float* row = onehot + (long)i * POS;
    float a = 0.f;
#pragma unroll
    for (int p = 0; p < POS; ++p) a += row[p] * Wp[p];
    out[i] = sigm(a + bp[0]);
}

// enc_cat[b,s,0:512]=out_f[b,s,:]; enc_cat[b,s,512:1024]=out_b[b,s,:]  (bf16)
__global__ void pack_enc_kernel(const float* __restrict__ of, const float* __restrict__ ob,
                                __bf16* __restrict__ cat) {
    int i = blockIdx.x * 256 + threadIdx.x;
    if (i >= BB * SS * 2 * HD) return;
    int j = i & 1023;
    int rs = i >> 10;
    float v = (j < HD) ? of[(long)rs * HD + j] : ob[(long)rs * HD + (j - HD)];
    cat[i] = f2bf(v);
}

// hcat = [hf|hb], ccat = [cf|cb]  bf16 (16,1024) each; hT/cT are (2,16,512) f32
__global__ void pack_hc_kernel(const float* __restrict__ hT, const float* __restrict__ cT,
                               __bf16* __restrict__ hcat, __bf16* __restrict__ ccat) {
    int i = blockIdx.x * 256 + threadIdx.x;
    if (i >= BB * 2 * HD) return;
    int b = i >> 10, j = i & 1023;
    long idx = (j < HD) ? ((long)b * HD + j) : ((long)BB * HD + (long)b * HD + (j - HD));
    hcat[i] = f2bf(hT[idx]);
    ccat[i] = f2bf(cT[idx]);
}

// ---------------------------------------------------------------------------
// Generic WMMA GEMM:  out(M,N) = act( A(M,K) @ W(N,K)^T + bias )
// block = 256 threads (8 waves); each wave computes a 16x64 tile.
// grid = (M/16, ceil((N/64)/8)).   M%16==0, N%64==0, K%32==0 required.
// ---------------------------------------------------------------------------
__global__ void __launch_bounds__(256)
gemm_bf16_wmma(const __bf16* __restrict__ A, const __bf16* __restrict__ W,
               const float* __restrict__ bias, float* __restrict__ outF,
               __bf16* __restrict__ outB, int M, int N, int K, int act) {
    int lane = threadIdx.x & 31;
    int wave = threadIdx.x >> 5;
    int m0 = blockIdx.x * 16;
    long n0 = ((long)blockIdx.y * 8 + wave) * 64;
    if (n0 >= N) return;                       // wave-uniform guard

    v8f acc[4];
#pragma unroll
    for (int u = 0; u < 4; ++u) acc[u] = vzero8();

    const __bf16* Arow = A + (long)m0 * K;
    for (int kb = 0; kb < K; kb += 32) {
        v16bf a = load_frag_A(Arow + kb, K, lane);
#pragma unroll
        for (int u = 0; u < 4; ++u) {
            v16bf b = load_frag_B(W + (n0 + u * 16) * (long)K + kb, K, lane);
            acc[u] = wmma_bf16(a, b, acc[u]);
        }
    }
    int mo = (lane >> 4) * 8;
#pragma unroll
    for (int u = 0; u < 4; ++u) {
        long n = n0 + u * 16 + (lane & 15);
#pragma unroll
        for (int r = 0; r < 8; ++r) {
            int m = m0 + mo + r;
            float v = acc[u][r];
            if (bias) v += bias[n];
            if (act == 1) v = FAST_TANH(v);
            long idx = (long)m * N + n;
            outF[idx] = v;
            if (outB) outB[idx] = f2bf(v);
        }
    }
}

// ---------------------------------------------------------------------------
// Encoder recurrence (one block per direction; 1024 thr = 32 waves).
// Per step: gates = h@Whh^T (WMMA) + xw[t] (precomputed x@Wih^T + biases),
// then masked LSTM cell update. h kept in LDS as bf16 (A-operand ready).
// Dynamic LDS: h bf16 16x512 | c f32 16x512 | gates f32 16x2048  = 176 KB
// ---------------------------------------------------------------------------
__global__ void __launch_bounds__(1024)
encoder_kernel(const __bf16* __restrict__ Whh_f, const __bf16* __restrict__ Whh_b,
               const float* __restrict__ xw_f, const float* __restrict__ xw_b,
               const int* __restrict__ lens,
               float* __restrict__ out_f, float* __restrict__ out_b,
               float* __restrict__ hT, float* __restrict__ cT) {
    extern __shared__ char smem[];
    __bf16* hS = (__bf16*)smem;                                   // 16*512 bf16
    float*  cS = (float*)(smem + BB * HD * 2);                    // 16*512 f32
    float*  gS = (float*)(smem + BB * HD * 2 + BB * HD * 4);      // 16*2048 f32

    int dir = blockIdx.x;
    const __bf16* Whh = dir ? Whh_b : Whh_f;
    const float*  xw  = dir ? xw_b  : xw_f;
    float*        out = dir ? out_b : out_f;

    int tid = threadIdx.x, lane = tid & 31, wave = tid >> 5;
    for (int q = tid; q < BB * HD; q += 1024) { hS[q] = f2bf(0.f); cS[q] = 0.f; }
    __syncthreads();

    int n0 = wave * 64;
    int mo = (lane >> 4) * 8;
    for (int t = 0; t < SS; ++t) {
        int s = dir ? (SS - 1 - t) : t;

        v8f acc[4];
#pragma unroll
        for (int u = 0; u < 4; ++u) acc[u] = vzero8();
        for (int kb = 0; kb < HD; kb += 32) {
            v16bf a = load_frag_A(hS + kb, HD, lane);
#pragma unroll
            for (int u = 0; u < 4; ++u) {
                v16bf b = load_frag_B(Whh + (long)(n0 + u * 16) * HD + kb, HD, lane);
                acc[u] = wmma_bf16(a, b, acc[u]);
            }
        }
#pragma unroll
        for (int u = 0; u < 4; ++u) {
            int n = n0 + u * 16 + (lane & 15);
#pragma unroll
            for (int r = 0; r < 8; ++r) {
                int m = mo + r;
                gS[m * G4 + n] = acc[u][r] + xw[((long)m * SS + s) * G4 + n];
            }
        }
        __syncthreads();

        for (int q = tid; q < BB * HD; q += 1024) {
            int m = q >> 9, j = q & 511;
            float gi = gS[m * G4 + j];
            float gf = gS[m * G4 + HD + j];
            float gg = gS[m * G4 + 2 * HD + j];
            float go = gS[m * G4 + 3 * HD + j];
            float c2 = sigm(gf) * cS[q] + sigm(gi) * FAST_TANH(gg);
            float h2 = sigm(go) * FAST_TANH(c2);
            bool mk = s < lens[m];
            if (mk) { cS[q] = c2; hS[q] = f2bf(h2); }
            out[((long)m * SS + s) * HD + j] = mk ? h2 : 0.f;
        }
        __syncthreads();
    }
    for (int q = tid; q < BB * HD; q += 1024) {
        hT[(long)dir * BB * HD + q] = (float)hS[q];
        cT[(long)dir * BB * HD + q] = cS[q];
    }
}

// ---------------------------------------------------------------------------
// Decoder recurrence (single block, 1024 thr = 32 waves, T=64 steps).
// Per step: hWs (WMMA) -> attention energy/softmax/ctx (VALU, f32x4 loads) ->
// gates = [emb|ctx]@Wih^T + h@Whh^T (WMMA chained) -> LSTM update.
// h_t stored f32 (hs) and bf16 (hsb) for the final FC GEMM.
// ---------------------------------------------------------------------------
__global__ void __launch_bounds__(1024)
decoder_kernel(const float* __restrict__ enc_out, const float* __restrict__ enc_proj,
               const float* __restrict__ gate_ps, const int* __restrict__ lens,
               const int* __restrict__ tgt, const float* __restrict__ dec_embed,
               const __bf16* __restrict__ Wih, const __bf16* __restrict__ Whh,
               const float* __restrict__ bias, const __bf16* __restrict__ Ws,
               const float* __restrict__ attnV, const float* __restrict__ h0,
               const float* __restrict__ c0,
               float* __restrict__ hs, __bf16* __restrict__ hsb) {
    extern __shared__ char smem[];
    char* p = smem;
    __bf16* hS   = (__bf16*)p;            p += BB * HD * 2;      // 16 KB
    float*  cS   = (float*)p;             p += BB * HD * 4;      // 32 KB
    float*  hws  = (float*)p;             p += BB * HD * 4;      // 32 KB
    float*  sc   = (float*)p;             p += BB * SS * 4;      // 16 KB
    __bf16* xcat = (__bf16*)p;            p += BB * XIN * 2;     // 20 KB
    float*  vS   = (float*)p;             p += HD * 4;           //  2 KB
    float*  gS   = (float*)p;                                    // 128 KB

    int tid = threadIdx.x, lane = tid & 31, wave = tid >> 5;
    int mo = (lane >> 4) * 8;

    for (int q = tid; q < BB * HD; q += 1024) { hS[q] = f2bf(h0[q]); cS[q] = c0[q]; }
    for (int q = tid; q < HD; q += 1024) vS[q] = attnV[q];
    __syncthreads();

    for (int t = 0; t < TT; ++t) {
        // ---- phase 1: hWs = h @ attn_Ws^T  (32 waves x one 16x16 tile)
        {
            int nw = wave * 16;
            v8f a1 = vzero8();
            for (int kb = 0; kb < HD; kb += 32) {
                v16bf a = load_frag_A(hS + kb, HD, lane);
                v16bf b = load_frag_B(Ws + (long)nw * HD + kb, HD, lane);
                a1 = wmma_bf16(a, b, a1);
            }
            int n = nw + (lane & 15);
#pragma unroll
            for (int r = 0; r < 8; ++r) hws[(mo + r) * HD + n] = a1[r];
        }
        __syncthreads();

        // ---- phase 2: scores[b,s] = gate * sum_h tanh(hWs + enc_proj) * V, masked
        for (int q = tid; q < BB * SS; q += 1024) {
            int b = q >> 8, si = q & 255;
            const float4* ep4 = (const float4*)(enc_proj + ((long)b * SS + si) * HD);
            const float4* hw4 = (const float4*)(hws + b * HD);
            const float4* v4  = (const float4*)vS;
            float a = 0.f;
            for (int h = 0; h < HD / 4; ++h) {
                float4 e = ep4[h], w = hw4[h], vv = v4[h];
                a += FAST_TANH(w.x + e.x) * vv.x;
                a += FAST_TANH(w.y + e.y) * vv.y;
                a += FAST_TANH(w.z + e.z) * vv.z;
                a += FAST_TANH(w.w + e.w) * vv.w;
            }
            a *= gate_ps[(long)b * SS + si];
            if (si >= lens[b]) a = -1e30f;
            sc[q] = a;
        }
        __syncthreads();

        // ---- phase 3: softmax per batch row (waves 0..15)
        if (wave < BB) {
            int b = wave;
            float mx = -1e30f;
            for (int s = lane; s < SS; s += 32) mx = fmaxf(mx, sc[b * SS + s]);
#pragma unroll
            for (int o = 16; o > 0; o >>= 1) mx = fmaxf(mx, __shfl_xor(mx, o, 32));
            float sum = 0.f;
            for (int s = lane; s < SS; s += 32) {
                float e = __expf(sc[b * SS + s] - mx);
                sc[b * SS + s] = e;
                sum += e;
            }
#pragma unroll
            for (int o = 16; o > 0; o >>= 1) sum += __shfl_xor(sum, o, 32);
            float inv = __builtin_amdgcn_rcpf(sum);
            for (int s = lane; s < SS; s += 32) sc[b * SS + s] *= inv;
        }
        __syncthreads();

        // ---- phase 4: ctx[b,h] = sum_s attn * enc_out  -> xcat[:,128:640] (f32x4 per thread)
        for (int q = tid; q < BB * (HD / 4); q += 1024) {
            int b = q >> 7, h4 = q & 127;     // h4 indexes groups of 4 h
            float4 a = {0.f, 0.f, 0.f, 0.f};
            const float4* eo = (const float4*)(enc_out + (long)b * SS * HD) + h4;
            const float*  aw = sc + b * SS;
            for (int s = 0; s < SS; ++s) {
                float w = aw[s];
                float4 e = eo[(long)s * (HD / 4)];
                a.x += w * e.x; a.y += w * e.y; a.z += w * e.z; a.w += w * e.w;
            }
            __bf16* xc = xcat + b * XIN + EMB + h4 * 4;
            xc[0] = f2bf(a.x); xc[1] = f2bf(a.y); xc[2] = f2bf(a.z); xc[3] = f2bf(a.w);
        }
        for (int q = tid; q < BB * EMB; q += 1024) {
            int b = q >> 7, e = q & 127;
            int tok = tgt[b * TT + t];
            xcat[b * XIN + e] = f2bf(dec_embed[(long)tok * EMB + e]);
        }
        __syncthreads();

        // ---- phase 5: gates = xcat @ Wih^T + h @ Whh^T + bias  (4 tiles / wave)
        {
            int n0 = wave * 64;
            v8f acc[4];
#pragma unroll
            for (int u = 0; u < 4; ++u) acc[u] = vzero8();
            for (int kb = 0; kb < XIN; kb += 32) {
                v16bf a = load_frag_A(xcat + kb, XIN, lane);
#pragma unroll
                for (int u = 0; u < 4; ++u) {
                    v16bf b = load_frag_B(Wih + (long)(n0 + u * 16) * XIN + kb, XIN, lane);
                    acc[u] = wmma_bf16(a, b, acc[u]);
                }
            }
            for (int kb = 0; kb < HD; kb += 32) {
                v16bf a = load_frag_A(hS + kb, HD, lane);
#pragma unroll
                for (int u = 0; u < 4; ++u) {
                    v16bf b = load_frag_B(Whh + (long)(n0 + u * 16) * HD + kb, HD, lane);
                    acc[u] = wmma_bf16(a, b, acc[u]);
                }
            }
#pragma unroll
            for (int u = 0; u < 4; ++u) {
                int n = n0 + u * 16 + (lane & 15);
#pragma unroll
                for (int r = 0; r < 8; ++r)
                    gS[(mo + r) * G4 + n] = acc[u][r] + bias[n];
            }
        }
        __syncthreads();

        // ---- phase 6: LSTM cell update, store h_t
        for (int q = tid; q < BB * HD; q += 1024) {
            int m = q >> 9, j = q & 511;
            float gi = gS[m * G4 + j];
            float gf = gS[m * G4 + HD + j];
            float gg = gS[m * G4 + 2 * HD + j];
            float go = gS[m * G4 + 3 * HD + j];
            float c2 = sigm(gf) * cS[q] + sigm(gi) * FAST_TANH(gg);
            float h2 = sigm(go) * FAST_TANH(c2);
            cS[q] = c2;
            hS[q] = f2bf(h2);
            long idx = ((long)m * TT + t) * HD + j;
            hs[idx]  = h2;
            hsb[idx] = f2bf(h2);
        }
        __syncthreads();
    }
}

// ---------------------------------------------------------------------------
// Host-side orchestration
// ---------------------------------------------------------------------------
extern "C" void kernel_launch(void* const* d_in, const int* in_sizes, int n_in,
                              void* d_out, int out_size, void* d_ws, size_t ws_size,
                              hipStream_t stream) {
    (void)in_sizes; (void)n_in; (void)out_size; (void)ws_size;
    const float* embedded   = (const float*)d_in[0];
    const int*   src_len    = (const int*)  d_in[1];
    const int*   tgt_tok    = (const int*)  d_in[2];
    const float* pos_onehot = (const float*)d_in[3];
    const float* eWih_f = (const float*)d_in[4];
    const float* eWhh_f = (const float*)d_in[5];
    const float* ebih_f = (const float*)d_in[6];
    const float* ebhh_f = (const float*)d_in[7];
    const float* eWih_b = (const float*)d_in[8];
    const float* eWhh_b = (const float*)d_in[9];
    const float* ebih_b = (const float*)d_in[10];
    const float* ebhh_b = (const float*)d_in[11];
    const float* poW  = (const float*)d_in[12];
    const float* pob  = (const float*)d_in[13];
    const float* phW  = (const float*)d_in[14];
    const float* phb  = (const float*)d_in[15];
    const float* pcW  = (const float*)d_in[16];
    const float* pcb  = (const float*)d_in[17];
    const float* aWs  = (const float*)d_in[18];
    const float* aWh  = (const float*)d_in[19];
    const float* aV   = (const float*)d_in[20];
    const float* aWp  = (const float*)d_in[21];
    const float* abp  = (const float*)d_in[22];
    const float* dEmb = (const float*)d_in[23];
    const float* dWih = (const float*)d_in[24];
    const float* dWhh = (const float*)d_in[25];
    const float* dbih = (const float*)d_in[26];
    const float* dbhh = (const float*)d_in[27];
    const float* fcW  = (const float*)d_in[28];
    const float* fcb  = (const float*)d_in[29];
    float* out = (float*)d_out;

    char* ws = (char*)d_ws;
    size_t off = 0;
    auto alloc = [&](size_t bytes) -> char* {
        char* q = ws + off;
        off += (bytes + 255) & ~(size_t)255;
        return q;
    };

    // bf16 buffers
    __bf16* emb_bf   = (__bf16*)alloc((size_t)BB * SS * INP * 2);
    __bf16* Wih_f_bf = (__bf16*)alloc((size_t)G4 * INP * 2);
    __bf16* Whh_f_bf = (__bf16*)alloc((size_t)G4 * HD * 2);
    __bf16* Wih_b_bf = (__bf16*)alloc((size_t)G4 * INP * 2);
    __bf16* Whh_b_bf = (__bf16*)alloc((size_t)G4 * HD * 2);
    __bf16* poW_bf   = (__bf16*)alloc((size_t)HD * 2 * HD * 2);
    __bf16* phW_bf   = (__bf16*)alloc((size_t)HD * 2 * HD * 2);
    __bf16* pcW_bf   = (__bf16*)alloc((size_t)HD * 2 * HD * 2);
    __bf16* aWs_bf   = (__bf16*)alloc((size_t)HD * HD * 2);
    __bf16* aWh_bf   = (__bf16*)alloc((size_t)HD * HD * 2);
    __bf16* dWih_bf  = (__bf16*)alloc((size_t)G4 * XIN * 2);
    __bf16* dWhh_bf  = (__bf16*)alloc((size_t)G4 * HD * 2);
    __bf16* fcW_bf   = (__bf16*)alloc((size_t)VOC * HD * 2);
    __bf16* enc_cat  = (__bf16*)alloc((size_t)BB * SS * 2 * HD * 2);
    __bf16* encout_bf= (__bf16*)alloc((size_t)BB * SS * HD * 2);
    __bf16* hs_bf    = (__bf16*)alloc((size_t)BB * TT * HD * 2);
    __bf16* hcat_bf  = (__bf16*)alloc((size_t)BB * 2 * HD * 2);
    __bf16* ccat_bf  = (__bf16*)alloc((size_t)BB * 2 * HD * 2);
    // f32 buffers
    float* xw_f    = (float*)alloc((size_t)BB * SS * G4 * 4);
    float* xw_b    = (float*)alloc((size_t)BB * SS * G4 * 4);
    float* out_f   = (float*)alloc((size_t)BB * SS * HD * 4);
    float* out_b   = (float*)alloc((size_t)BB * SS * HD * 4);
    float* hT      = (float*)alloc((size_t)2 * BB * HD * 4);
    float* cT      = (float*)alloc((size_t)2 * BB * HD * 4);
    float* enc_out = (float*)alloc((size_t)BB * SS * HD * 4);
    float* enc_prj = (float*)alloc((size_t)BB * SS * HD * 4);
    float* h0      = (float*)alloc((size_t)BB * HD * 4);
    float* c0      = (float*)alloc((size_t)BB * HD * 4);
    float* gate_ps = (float*)alloc((size_t)BB * SS * 4);
    float* bsum_f  = (float*)alloc((size_t)G4 * 4);
    float* bsum_b  = (float*)alloc((size_t)G4 * 4);
    float* bsum_d  = (float*)alloc((size_t)G4 * 4);
    float* hs      = (float*)alloc((size_t)BB * TT * HD * 4);

    auto cvt = [&](const float* src, __bf16* dst, int n) {
        cvt_bf16_kernel<<<(n + 255) / 256, 256, 0, stream>>>(src, dst, n);
    };

    // 1) convert weights / activations to bf16
    cvt(embedded, emb_bf, BB * SS * INP);
    cvt(eWih_f, Wih_f_bf, G4 * INP);
    cvt(eWhh_f, Whh_f_bf, G4 * HD);
    cvt(eWih_b, Wih_b_bf, G4 * INP);
    cvt(eWhh_b, Whh_b_bf, G4 * HD);
    cvt(poW, poW_bf, HD * 2 * HD);
    cvt(phW, phW_bf, HD * 2 * HD);
    cvt(pcW, pcW_bf, HD * 2 * HD);
    cvt(aWs, aWs_bf, HD * HD);
    cvt(aWh, aWh_bf, HD * HD);
    cvt(dWih, dWih_bf, G4 * XIN);
    cvt(dWhh, dWhh_bf, G4 * HD);
    cvt(fcW, fcW_bf, VOC * HD);

    // 2) bias sums
    bias_sum_kernel<<<(G4 + 255) / 256, 256, 0, stream>>>(ebih_f, ebhh_f, bsum_f, G4);
    bias_sum_kernel<<<(G4 + 255) / 256, 256, 0, stream>>>(ebih_b, ebhh_b, bsum_b, G4);
    bias_sum_kernel<<<(G4 + 255) / 256, 256, 0, stream>>>(dbih, dbhh, bsum_d, G4);

    // 3) xw = x @ Wih^T + (bih+bhh)   both directions  (M=4096,N=2048,K=128)
    {
        dim3 grid(BB * SS / 16, (G4 / 64 + 7) / 8);
        gemm_bf16_wmma<<<grid, 256, 0, stream>>>(emb_bf, Wih_f_bf, bsum_f, xw_f, nullptr,
                                                 BB * SS, G4, INP, 0);
        gemm_bf16_wmma<<<grid, 256, 0, stream>>>(emb_bf, Wih_b_bf, bsum_b, xw_b, nullptr,
                                                 BB * SS, G4, INP, 0);
    }

    // 4) encoder recurrence (block 0 = fwd, block 1 = bwd)
    {
        size_t shm = (size_t)BB * HD * 2 + (size_t)BB * HD * 4 + (size_t)BB * G4 * 4; // 176 KB
        encoder_kernel<<<2, 1024, shm, stream>>>(Whh_f_bf, Whh_b_bf, xw_f, xw_b, src_len,
                                                 out_f, out_b, hT, cT);
    }

    // 5) enc_cat = concat(out_f, out_b)  (bf16)
    pack_enc_kernel<<<(BB * SS * 2 * HD + 255) / 256, 256, 0, stream>>>(out_f, out_b, enc_cat);

    // 6) enc_out = enc_cat @ proj_out_W^T + b   (M=4096,N=512,K=1024)
    {
        dim3 grid(BB * SS / 16, 1);
        gemm_bf16_wmma<<<grid, 256, 0, stream>>>(enc_cat, poW_bf, pob, enc_out, encout_bf,
                                                 BB * SS, HD, 2 * HD, 0);
        // 7) enc_proj = enc_out @ attn_Wh^T     (M=4096,N=512,K=512)
        gemm_bf16_wmma<<<grid, 256, 0, stream>>>(encout_bf, aWh_bf, nullptr, enc_prj, nullptr,
                                                 BB * SS, HD, HD, 0);
    }

    // 8) h0/c0 = tanh(concat @ W^T + b)
    pack_hc_kernel<<<(BB * 2 * HD + 255) / 256, 256, 0, stream>>>(hT, cT, hcat_bf, ccat_bf);
    {
        dim3 grid(1, 1);
        gemm_bf16_wmma<<<grid, 256, 0, stream>>>(hcat_bf, phW_bf, phb, h0, nullptr,
                                                 BB, HD, 2 * HD, 1);
        gemm_bf16_wmma<<<grid, 256, 0, stream>>>(ccat_bf, pcW_bf, pcb, c0, nullptr,
                                                 BB, HD, 2 * HD, 1);
    }

    // 9) position gate
    gate_ps_kernel<<<(BB * SS + 255) / 256, 256, 0, stream>>>(pos_onehot, aWp, abp, gate_ps);

    // 10) decoder recurrence
    {
        size_t shm = (size_t)BB * HD * 2 + (size_t)BB * HD * 4 + (size_t)BB * HD * 4 +
                     (size_t)BB * SS * 4 + (size_t)BB * XIN * 2 + (size_t)HD * 4 +
                     (size_t)BB * G4 * 4;                                   // ~246 KB
        decoder_kernel<<<1, 1024, shm, stream>>>(enc_out, enc_prj, gate_ps, src_len, tgt_tok,
                                                 dEmb, dWih_bf, dWhh_bf, bsum_d, aWs_bf, aV,
                                                 h0, c0, hs, hs_bf);
    }

    // 11) preds = hs @ fc_W^T + fc_b   (M=1024,N=32000,K=512) -> d_out (B,T,VOCAB)
    {
        dim3 grid(BB * TT / 16, (VOC / 64 + 7) / 8);
        gemm_bf16_wmma<<<grid, 256, 0, stream>>>(hs_bf, fcW_bf, fcb, out, nullptr,
                                                 BB * TT, VOC, HD, 0);
    }
}
